// MultiHeadCausalAttention_63668595196336
// MI455X (gfx1250) — matrix operand order repaired and checked by reading, not compiled, not run on hardware
//
#include <hip/hip_runtime.h>

#define DI __device__ __forceinline__
#define USE_ASYNC_LDS 1

typedef __attribute__((ext_vector_type(16))) __bf16 bf16x16;
typedef __attribute__((ext_vector_type(8)))  float  v8f;

static constexpr int B_  = 4;
static constexpr int T_  = 2048;
static constexpr int D_  = 1024;
static constexpr int H_  = 16;
static constexpr int HD_ = 64;
static constexpr int BT_ = B_ * T_;          // 8192 rows

// 16x32 bf16 A-frag / 32x16 bf16 B-frag: 32 bytes per lane.
union ABfrag {
  bf16x16        v;
  uint4          q[2];
  unsigned short s[16];
};

DI unsigned short f2bf(float f) {
  unsigned u = __float_as_uint(f);
  u += 0x7FFFu + ((u >> 16) & 1u);           // round-to-nearest-even
  return (unsigned short)(u >> 16);
}

// Async global->LDS copy, 16B per lane (ASYNCcnt path).
DI void async_ld16(unsigned short* lds_dst, const unsigned short* gsrc) {
#if USE_ASYNC_LDS
  unsigned loff = (unsigned)(unsigned long long)lds_dst;   // low 32b = LDS offset
  asm volatile("global_load_async_to_lds_b128 %0, %1, off"
               :: "v"(loff), "v"(gsrc) : "memory");
#else
  *(uint4*)lds_dst = *(const uint4*)gsrc;
#endif
}
DI void async_wait() {
#if USE_ASYNC_LDS
  asm volatile("s_wait_asynccnt 0" ::: "memory");
#endif
}

// ---------------------------------------------------------------- converts
__global__ __launch_bounds__(256) void cvt_f32_bf16(
    const float* __restrict__ in, unsigned short* __restrict__ out, int n) {
  int i = (blockIdx.x * 256 + threadIdx.x) * 4;
  if (i + 3 < n) {
    float4 x = *(const float4*)(in + i);
    unsigned lo = (unsigned)f2bf(x.x) | ((unsigned)f2bf(x.y) << 16);
    unsigned hi = (unsigned)f2bf(x.z) | ((unsigned)f2bf(x.w) << 16);
    *(uint2*)(out + i) = make_uint2(lo, hi);
  }
}

// ------------------------------------------------------------ GEMM helpers
// Wave tile: 32(M) x 64(N). 2 A-frags + 4 B-frags -> 8 wmmas per 32-K step.
DI void load_ab2(ABfrag a[2], ABfrag b[4], const unsigned short* aRow0,
                 const unsigned short* wRow, int kk) {
  a[0].q[0] = *(const uint4*)(aRow0 + kk);
  a[0].q[1] = *(const uint4*)(aRow0 + kk + 16);
  a[1].q[0] = *(const uint4*)(aRow0 + 16 * D_ + kk);
  a[1].q[1] = *(const uint4*)(aRow0 + 16 * D_ + kk + 16);
#pragma unroll
  for (int s = 0; s < 4; ++s) {
    const unsigned short* bRow = wRow + s * (16 * D_) + kk;
    b[s].q[0] = *(const uint4*)(bRow);
    b[s].q[1] = *(const uint4*)(bRow + 16);
  }
}

DI void wmma8(v8f acc[2][4], const ABfrag a[2], const ABfrag b[4]) {
#pragma unroll
  for (int s = 0; s < 4; ++s)
#pragma unroll
    for (int m = 0; m < 2; ++m)
      acc[m][s] = __builtin_amdgcn_wmma_f32_16x16x32_bf16(
          false, a[m].v, false, b[s].v, (short)0, acc[m][s], false, false);
}

// Register double-buffered K loop: loads for k+32 issue before wmmas of k.
DI void gemm_k_loop(v8f acc[2][4], const unsigned short* aRow0,
                    const unsigned short* wRow) {
  ABfrag a[2][2], b[2][4];
  load_ab2(a[0], b[0], aRow0, wRow, 0);
  for (int kk = 0; kk < D_; kk += 64) {
    __builtin_prefetch(aRow0 + kk + 512, 0, 1);
    __builtin_prefetch(wRow + kk + 512, 0, 1);
    if (kk + 32 < D_) load_ab2(a[1], b[1], aRow0, wRow, kk + 32);
    wmma8(acc, a[0], b[0]);
    if (kk + 64 < D_) load_ab2(a[0], b[0], aRow0, wRow, kk + 64);
    wmma8(acc, a[1], b[1]);
  }
}

// ------------------------------------------------------ QKV projection GEMM
// out = x @ W^T. A = xb [BT, D] bf16, B-col n = W row n (contiguous).
// mode 0/1: write [B,H,T,HD]; mode 2: write V transposed [B,H,HD,T].
__global__ __launch_bounds__(256)
__attribute__((amdgpu_waves_per_eu(2))) void qkv_gemm(
    const unsigned short* __restrict__ xb,
    const unsigned short* __restrict__ wb,
    unsigned short* __restrict__ outp, int mode) {
  const int lane = threadIdx.x & 31;
  const int wave = threadIdx.x >> 5;
  const int gw   = blockIdx.x * 8 + wave;    // 4096 wave tiles
  const int mBase = (gw >> 4) << 5;          // 256 M-tiles of 32
  const int nBase = (gw & 15) << 6;          // 16 N-groups of 64
  const int lr = lane & 15, lh = lane >> 4;

  v8f acc[2][4] = {};
  const unsigned short* aRow0 = xb + (mBase + lr) * D_ + (lh << 3);
  const unsigned short* wRow  = wb + (nBase + lr) * D_ + (lh << 3);
  gemm_k_loop(acc, aRow0, wRow);

#pragma unroll
  for (int ms = 0; ms < 2; ++ms)
#pragma unroll
    for (int s = 0; s < 4; ++s) {
      int n = nBase + (s << 4) + lr;
      int h = n >> 6, hd = n & 63;
#pragma unroll
      for (int i = 0; i < 8; ++i) {
        int m = mBase + (ms << 4) + (lh << 3) + i;
        int b = m >> 11, t = m & (T_ - 1);
        int idx;
        if (mode == 2) idx = ((b * H_ + h) * HD_ + hd) * T_ + t;   // V^T
        else           idx = ((b * H_ + h) * T_ + t) * HD_ + hd;   // Q,K
        outp[idx] = f2bf(acc[ms][s][i]);
      }
    }
}

// ----------------------------------------------------------- flash attention
// Block = 8 q-tiles (128 queries) of one (b,h). K/V tiles double-buffered in
// LDS: async copies for block k+1 issue before the compute of block k.
__global__ __launch_bounds__(256) void flash_attn(
    const unsigned short* __restrict__ qb,   // [B,H,T,HD]
    const unsigned short* __restrict__ kb,   // [B,H,T,HD]
    const unsigned short* __restrict__ vtb,  // [B,H,HD,T]
    unsigned short* __restrict__ ao) {       // [BT, D]
  __shared__ __align__(16) unsigned short kt[2][32 * 64];     // K [key][d]
  __shared__ __align__(16) unsigned short vtile[2][64 * 32];  // V [hd][key]
  __shared__ __align__(16) unsigned short psh[8][16 * 32];    // per-wave P

  const int tid  = threadIdx.x;
  const int lane = tid & 31;
  const int wave = tid >> 5;
  const int bh     = blockIdx.x >> 4;        // 64 (b,h) pairs
  const int qGroup = blockIdx.x & 15;        // 16 groups of 128 queries
  const int qBase  = (qGroup << 7) + (wave << 4);
  const int lr = lane & 15, lh = lane >> 4;

  const unsigned short* qhead = qb  + bh * (T_ * HD_);
  const unsigned short* khead = kb  + bh * (T_ * HD_);
  const unsigned short* vhead = vtb + bh * (HD_ * T_);

  // staging coordinates (256 threads move 8KB per key block)
  const int krow = tid >> 3, kcol = (tid & 7) << 3;   // 32 x (8x8)
  const int vrow = tid >> 2, vcol = (tid & 3) << 3;   // 64 x (4x8)
  auto stage = [&](int buf, int kBase) {
    async_ld16(&kt[buf][krow * 64 + kcol], khead + (kBase + krow) * HD_ + kcol);
    async_ld16(&vtile[buf][vrow * 32 + vcol], vhead + vrow * T_ + kBase + vcol);
  };

  ABfrag aq[2];                               // Q 16x64 = 2 A-frags over d
  {
    const unsigned short* qrow = qhead + (qBase + lr) * HD_ + (lh << 4);
    aq[0].q[0] = *(const uint4*)(qrow);
    aq[0].q[1] = *(const uint4*)(qrow + 8);
    aq[1].q[0] = *(const uint4*)(qrow + 32);
    aq[1].q[1] = *(const uint4*)(qrow + 40);
  }

  v8f o[4] = {};
  float mrun[8], lrun[8];
#pragma unroll
  for (int i = 0; i < 8; ++i) { mrun[i] = -1e30f; lrun[i] = 0.f; }

  const int myNkb  = (qBase + 16 + 31) >> 5;   // this wave's causal extent
  const int nkbMax = (qGroup << 2) + 4;        // block-uniform trip count

  stage(0, 0);
  async_wait();
  __syncthreads();

  for (int kblk = 0; kblk < nkbMax; ++kblk) {
    const int cur = kblk & 1;
    const int kBase = kblk << 5;
    if (kblk + 1 < nkbMax) stage(cur ^ 1, (kblk + 1) << 5);  // overlap copy

    if (kblk < myNkb) {                        // wave-uniform branch
      const unsigned short* ktc = kt[cur];
      const unsigned short* vtc = vtile[cur];
      v8f sfr[2];
#pragma unroll
      for (int kn = 0; kn < 2; ++kn) {         // S = Q K^T, 16 keys per frag
        const unsigned short* krw = ktc + ((kn << 4) + lr) * 64 + (lh << 4);
        ABfrag b0, b1;
        b0.q[0] = *(const uint4*)(krw);        // d = lh*16 .. +15
        b0.q[1] = *(const uint4*)(krw + 8);
        b1.q[0] = *(const uint4*)(krw + 32);   // d = 32 + lh*16 .. +15
        b1.q[1] = *(const uint4*)(krw + 40);
        v8f z = {};
        z = __builtin_amdgcn_wmma_f32_16x16x32_bf16(false, aq[0].v, false, b0.v, (short)0, z, false, false);
        z = __builtin_amdgcn_wmma_f32_16x16x32_bf16(false, aq[1].v, false, b1.v, (short)0, z, false, false);
        sfr[kn] = z;
      }

      // scale, causal mask, block row-max (rows 0-7 / 8-15 by lane half)
      float mblk[8];
#pragma unroll
      for (int i = 0; i < 8; ++i) {
        int qi = qBase + (lh << 3) + i;
        float s0 = sfr[0][i] * 0.125f; if (kBase + lr > qi)      s0 = -1e30f;
        float s1 = sfr[1][i] * 0.125f; if (kBase + 16 + lr > qi) s1 = -1e30f;
        sfr[0][i] = s0; sfr[1][i] = s1;
        mblk[i] = fmaxf(s0, s1);
      }
#pragma unroll
      for (int d = 1; d < 16; d <<= 1)
#pragma unroll
        for (int i = 0; i < 8; ++i) mblk[i] = fmaxf(mblk[i], __shfl_xor(mblk[i], d, 32));

      float alpha[8], lblk[8];
#pragma unroll
      for (int i = 0; i < 8; ++i) {
        float mn = fmaxf(mrun[i], mblk[i]);
        alpha[i] = __expf(mrun[i] - mn);
        mrun[i] = mn;
        float p0 = __expf(sfr[0][i] - mn);
        float p1 = __expf(sfr[1][i] - mn);
        sfr[0][i] = p0; sfr[1][i] = p1;
        lblk[i] = p0 + p1;
      }
#pragma unroll
      for (int d = 1; d < 16; d <<= 1)
#pragma unroll
        for (int i = 0; i < 8; ++i) lblk[i] += __shfl_xor(lblk[i], d, 32);
#pragma unroll
      for (int i = 0; i < 8; ++i) {
        lrun[i] = lrun[i] * alpha[i] + lblk[i];
#pragma unroll
        for (int s = 0; s < 4; ++s) o[s][i] *= alpha[i];
      }

      // transpose P (C-layout -> A-layout) through per-wave LDS tile
      unsigned short* P = psh[wave];
#pragma unroll
      for (int i = 0; i < 8; ++i) {
        int r = (lh << 3) + i;
        P[r * 32 + lr]      = f2bf(sfr[0][i]);
        P[r * 32 + 16 + lr] = f2bf(sfr[1][i]);
      }
      asm volatile("s_wait_dscnt 0" ::: "memory");   // same-wave LDS RAW
      ABfrag pa;
      pa.q[0] = *(const uint4*)(P + lr * 32 + (lh << 3));
      pa.q[1] = *(const uint4*)(P + lr * 32 + 16 + (lh << 3));

      // O += P @ V from LDS V tile
#pragma unroll
      for (int s = 0; s < 4; ++s) {
        const unsigned short* vrw = vtc + ((s << 4) + lr) * 32 + (lh << 3);
        ABfrag bv;
        bv.q[0] = *(const uint4*)(vrw);
        bv.q[1] = *(const uint4*)(vrw + 16);
        o[s] = __builtin_amdgcn_wmma_f32_16x16x32_bf16(
            false, pa.v, false, bv.v, (short)0, o[s], false, false);
      }
    }
    async_wait();                              // staged copies done
    __syncthreads();                           // all readers done with cur
  }

  const int h = bh & (H_ - 1), b = bh >> 4;
#pragma unroll
  for (int s = 0; s < 4; ++s) {
    int col = h * HD_ + (s << 4) + lr;
#pragma unroll
    for (int i = 0; i < 8; ++i) {
      int m = b * T_ + qBase + (lh << 3) + i;
      ao[m * D_ + col] = f2bf(o[s][i] / lrun[i]);
    }
  }
}

// ------------------------------------------- output projection + residual
__global__ __launch_bounds__(256)
__attribute__((amdgpu_waves_per_eu(2))) void out_proj(
    const unsigned short* __restrict__ ao,   // [BT, D] bf16
    const unsigned short* __restrict__ wob,  // [D, D] bf16
    const float* __restrict__ x,             // residual
    float* __restrict__ y) {
  const int lane = threadIdx.x & 31;
  const int wave = threadIdx.x >> 5;
  const int gw   = blockIdx.x * 8 + wave;
  const int mBase = (gw >> 4) << 5;
  const int nBase = (gw & 15) << 6;
  const int lr = lane & 15, lh = lane >> 4;

  v8f acc[2][4] = {};
  const unsigned short* aRow0 = ao + (mBase + lr) * D_ + (lh << 3);
  const unsigned short* wRow  = wob + (nBase + lr) * D_ + (lh << 3);
  gemm_k_loop(acc, aRow0, wRow);

#pragma unroll
  for (int ms = 0; ms < 2; ++ms)
#pragma unroll
    for (int s = 0; s < 4; ++s) {
      int n = nBase + (s << 4) + lr;
#pragma unroll
      for (int i = 0; i < 8; ++i) {
        int m = mBase + (ms << 4) + (lh << 3) + i;
        y[m * D_ + n] = x[m * D_ + n] + acc[ms][s][i];
      }
    }
}

// ----------------------------------------------------------------- RMSNorm
__global__ __launch_bounds__(256) void rmsnorm(
    const float* __restrict__ y, const float* __restrict__ g,
    float* __restrict__ out) {
  __shared__ float red[8];
  const int row = blockIdx.x;
  const float* yr = y + row * D_;
  float4 v = *(const float4*)(yr + threadIdx.x * 4);
  float ss = v.x * v.x + v.y * v.y + v.z * v.z + v.w * v.w;
#pragma unroll
  for (int d = 1; d < 32; d <<= 1) ss += __shfl_xor(ss, d, 32);
  if ((threadIdx.x & 31) == 0) red[threadIdx.x >> 5] = ss;
  __syncthreads();
  float tot = 0.f;
#pragma unroll
  for (int i = 0; i < 8; ++i) tot += red[i];
  float inv = rsqrtf(tot * (1.0f / D_) + 1e-6f);
  float4 gg = *(const float4*)(g + threadIdx.x * 4);
  float4 o;
  o.x = v.x * inv * gg.x; o.y = v.y * inv * gg.y;
  o.z = v.z * inv * gg.z; o.w = v.w * inv * gg.w;
  *(float4*)(out + row * D_ + threadIdx.x * 4) = o;
}

// ---------------------------------------------------------------- launcher
extern "C" void kernel_launch(void* const* d_in, const int* in_sizes, int n_in,
                              void* d_out, int out_size, void* d_ws, size_t ws_size,
                              hipStream_t stream) {
  const float* x  = (const float*)d_in[0];
  const float* wq = (const float*)d_in[1];
  const float* wk = (const float*)d_in[2];
  const float* wv = (const float*)d_in[3];
  const float* wo = (const float*)d_in[4];
  const float* g  = (const float*)d_in[5];

  char* ws = (char*)d_ws;
  unsigned short* xb  = (unsigned short*)(ws);                     // 16 MB
  unsigned short* wqb = (unsigned short*)(ws + (size_t)16777216);  // 2 MB each
  unsigned short* wkb = wqb + 1048576;
  unsigned short* wvb = wkb + 1048576;
  unsigned short* wob = wvb + 1048576;
  unsigned short* q   = (unsigned short*)(ws + (size_t)25165824);  // 16 MB each
  unsigned short* k   = q  + 8388608;
  unsigned short* vt  = k  + 8388608;
  unsigned short* ao  = vt + 8388608;
  float*          y   = (float*)(ws + (size_t)92274688);           // 32 MB
  float*          out = (float*)d_out;

  cvt_f32_bf16<<<8192, 256, 0, stream>>>(x,  xb,  BT_ * D_);
  cvt_f32_bf16<<<1024, 256, 0, stream>>>(wq, wqb, D_ * D_);
  cvt_f32_bf16<<<1024, 256, 0, stream>>>(wk, wkb, D_ * D_);
  cvt_f32_bf16<<<1024, 256, 0, stream>>>(wv, wvb, D_ * D_);
  cvt_f32_bf16<<<1024, 256, 0, stream>>>(wo, wob, D_ * D_);

  qkv_gemm<<<512, 256, 0, stream>>>(xb, wqb, q,  0);
  qkv_gemm<<<512, 256, 0, stream>>>(xb, wkb, k,  1);
  qkv_gemm<<<512, 256, 0, stream>>>(xb, wvb, vt, 2);

  flash_attn<<<1024, 256, 0, stream>>>(q, k, vt, ao);
  out_proj<<<512, 256, 0, stream>>>(ao, wob, x, y);
  rmsnorm<<<8192, 256, 0, stream>>>(y, g, out);
}
